// EncoderLayer_39608188404139
// MI455X (gfx1250) — matrix operand order repaired and checked by reading, not compiled, not run on hardware
//
#include <hip/hip_runtime.h>
#include <cstdint>
#include <cstddef>

// ---------------------------------------------------------------------------
// MI455X / gfx1250: wave32, WMMA 16x16x32 bf16 (f32 accumulate).
// GEMMs are register-blocked 64x64 per wave (4x4 WMMA tiles) for 4x higher
// arithmetic intensity on the compute-bound matrix path.
// ---------------------------------------------------------------------------

#define DEV __device__ __forceinline__

typedef __attribute__((ext_vector_type(16))) __bf16 v16bf;
typedef __attribute__((ext_vector_type(8)))  float  v8f;
typedef __attribute__((ext_vector_type(4)))  unsigned u32x4;   // trivially-ctor 16B vector
typedef __attribute__((ext_vector_type(4)))  float    f32x4;

union ABf { u32x4 u[2]; v16bf v; };   // 32 bytes = 16 bf16 (one WMMA operand fragment)

constexpr int Bb   = 256;
constexpr int Ss   = 64;
constexpr int Dd   = 1024;
constexpr int Hh   = 32;
constexpr int DHd  = 32;
constexpr int Ee   = 4;
constexpr int HIDh = 1536;
constexpr int CAP  = 32;            // max(min(64, 64*2/4), 4)
constexpr int Mtok = Bb * Ss;       // 16384
constexpr int EROW = Bb * CAP;      // 8192 rows per expert

DEV __bf16 f2bf(float f) {
  unsigned u = __builtin_bit_cast(unsigned, f);
  unsigned r = u + 0x7FFFu + ((u >> 16) & 1u);      // round-to-nearest-even
  unsigned short h = (unsigned short)(r >> 16);
  return __builtin_bit_cast(__bf16, h);
}
DEV float bf2f(__bf16 b) {
  unsigned short h = __builtin_bit_cast(unsigned short, b);
  return __builtin_bit_cast(float, (unsigned)h << 16);
}
DEV v8f wmma_bf16(v16bf a, v16bf b, v8f c) {
  return __builtin_amdgcn_wmma_f32_16x16x32_bf16(false, a, false, b, (short)0, c,
                                                 false, false);
}
DEV v8f v8zero() {
  v8f z = {0.f, 0.f, 0.f, 0.f, 0.f, 0.f, 0.f, 0.f};
  return z;
}
DEV int imin(int a, int b) { return a < b ? a : b; }

// ---------------------------------------------------------------------------
// elementwise converts / transposes into bf16
// ---------------------------------------------------------------------------
__global__ __launch_bounds__(256) void k_f32_to_bf16(const float* __restrict__ in,
                                                     __bf16* __restrict__ out, int n) {
  int i = blockIdx.x * 256 + threadIdx.x;
  if (i < n) out[i] = f2bf(in[i]);
}

__global__ __launch_bounds__(256) void k_transpose_bf16(const float* __restrict__ in,
                                                        __bf16* __restrict__ out,
                                                        int R, int C) {
  int i = blockIdx.x * 256 + threadIdx.x;
  if (i >= R * C) return;
  int r = i / C, c = i % C;
  out[(size_t)c * R + r] = f2bf(in[i]);   // out[C,R] = in[R,C]^T
}

// ---------------------------------------------------------------------------
// Register-blocked WMMA GEMM: C[M,N] = A[M,K] (bf16 row-major) @ Bt[N,K]^T.
// One 64x64 macro-tile per wave = 4x4 grid of 16x16 WMMA tiles.
// Per k-step(32): 8 fragment loads (4 A + 4 B, 2x b128 each) feed 16 WMMAs.
// MODE selects epilogue (fused bias / layout scatter / relu).
// ---------------------------------------------------------------------------
template <int MODE>
__global__ __launch_bounds__(256) void k_gemm(
    const __bf16* __restrict__ A, const __bf16* __restrict__ Bt,
    const float* __restrict__ bias, const float* __restrict__ ubias,
    const float* __restrict__ vbias, void* __restrict__ out0,
    void* __restrict__ out1, int M, int N, int K) {
  const int wave = threadIdx.x >> 5, lane = threadIdx.x & 31;
  const int l15 = lane & 15, hh = lane >> 4;
  const int ntm = N >> 6;                               // 64-wide macro columns
  const int tile = blockIdx.x * 8 + wave;
  if (tile >= (M >> 6) * ntm) return;                   // wave-uniform guard
  const int tm = tile / ntm, tn = tile % ntm;
  const int m0 = tm << 6, n0 = tn << 6;

  // A fragment: lane row (m0+16i+l15), K runs [8h..8h+7] and [16+8h..16+8h+7]
  const __bf16* Ar = A + (size_t)(m0 + l15) * K + 8 * hh;
  // B fragment: lane col (n0+16j+l15), K runs [8h..8h+7] and [16+8h..16+8h+7]
  const __bf16* Br = Bt + (size_t)(n0 + l15) * K + 8 * hh;

  v8f acc[4][4];
#pragma unroll
  for (int i = 0; i < 4; ++i)
#pragma unroll
    for (int j = 0; j < 4; ++j) acc[i][j] = v8zero();

  for (int kb = 0; kb < K; kb += 32) {
    ABf af[4], bf[4];
#pragma unroll
    for (int i = 0; i < 4; ++i) {
      const __bf16* pa = Ar + (size_t)(16 * i) * K + kb;
      const __bf16* pb = Br + (size_t)(16 * i) * K + kb;
      af[i].u[0] = *(const u32x4*)(pa);
      af[i].u[1] = *(const u32x4*)(pa + 16);
      bf[i].u[0] = *(const u32x4*)(pb);
      bf[i].u[1] = *(const u32x4*)(pb + 16);
    }
    __builtin_prefetch(Ar + kb + 256, 0, 1);            // global_prefetch_b8
    __builtin_prefetch(Br + kb + 256, 0, 1);
#pragma unroll
    for (int i = 0; i < 4; ++i)
#pragma unroll
      for (int j = 0; j < 4; ++j)
        acc[i][j] = wmma_bf16(af[i].v, bf[j].v, acc[i][j]);
  }

#pragma unroll
  for (int i = 0; i < 4; ++i) {
#pragma unroll
    for (int j = 0; j < 4; ++j) {
      const int n = n0 + 16 * j + l15;
#pragma unroll
      for (int r = 0; r < 8; ++r) {
        const int m = m0 + 16 * i + r + 8 * hh;         // C layout: row r+8h, col l15
        float val = acc[i][j][r];
        if constexpr (MODE == 1) {                      // Q -> qu,qv [b,h,s,dh]
          val += bias[n];
          const int b = m >> 6, s = m & 63, h = n >> 5, dh = n & 31;
          const size_t idx = (((size_t)(b * Hh + h)) * Ss + s) * DHd + dh;
          ((__bf16*)out0)[idx] = f2bf(val + ubias[n]);
          ((__bf16*)out1)[idx] = f2bf(val + vbias[n]);
        } else if constexpr (MODE == 2) {               // K -> [b,h,s,dh]
          val += bias[n];
          const int b = m >> 6, s = m & 63, h = n >> 5, dh = n & 31;
          ((__bf16*)out0)[(((size_t)(b * Hh + h)) * Ss + s) * DHd + dh] = f2bf(val);
        } else if constexpr (MODE == 3) {               // V -> [b,h,dh,s] (pre-transposed)
          val += bias[n];
          const int b = m >> 6, s = m & 63, h = n >> 5, dh = n & 31;
          ((__bf16*)out0)[(((size_t)(b * Hh + h)) * DHd + dh) * Ss + s] = f2bf(val);
        } else if constexpr (MODE == 4) {               // pos -> [h,t,dh], no bias
          const int h = n >> 5, dh = n & 31;
          ((__bf16*)out0)[((size_t)h * Ss + m) * DHd + dh] = f2bf(val);
        } else if constexpr (MODE == 5) {               // Wo -> f32 [M,N] + bias
          ((float*)out0)[(size_t)m * N + n] = val + bias[n];
        } else if constexpr (MODE == 6) {               // expert w1 -> relu, bf16
          ((__bf16*)out0)[(size_t)m * N + n] = f2bf(fmaxf(val, 0.f));
        } else {                                        // MODE 7: expert w2 -> bf16
          ((__bf16*)out0)[(size_t)m * N + n] = f2bf(val);
        }
      }
    }
  }
}

// ---------------------------------------------------------------------------
// Fused attention per (b,h): content & pos score WMMAs -> LDS, rel-shift,
// softmax, ctx WMMA (attn read from LDS), ctx scattered to [token, D].
// ---------------------------------------------------------------------------
__global__ __launch_bounds__(512) void k_attn(
    const __bf16* __restrict__ qu, const __bf16* __restrict__ qv,
    const __bf16* __restrict__ kmat, const __bf16* __restrict__ vt,
    const __bf16* __restrict__ posb, __bf16* __restrict__ ctxb) {
  __shared__ float sc[64][66];
  __shared__ float ps[64][66];
  const int bh = blockIdx.x, b = bh >> 5, h = bh & 31;
  const int wave = threadIdx.x >> 5, lane = threadIdx.x & 31;
  const int l15 = lane & 15, hh = lane >> 4;

  {  // 16 waves: one 16x16 score tile each (content + positional), K = DH = 32
    const int tm = wave >> 2, tn = wave & 3;
    const __bf16* Aq = qu   + ((size_t)bh * Ss + tm * 16 + l15) * DHd + 8 * hh;
    const __bf16* Av = qv   + ((size_t)bh * Ss + tm * 16 + l15) * DHd + 8 * hh;
    const __bf16* Bk = kmat + ((size_t)bh * Ss + tn * 16 + l15) * DHd + 8 * hh;
    const __bf16* Bp = posb + ((size_t)h  * Ss + tn * 16 + l15) * DHd + 8 * hh;
    ABf a, bm;
    v8f acc = v8zero();
    a.u[0] = *(const u32x4*)(Aq);      a.u[1] = *(const u32x4*)(Aq + 16);
    bm.u[0] = *(const u32x4*)(Bk);     bm.u[1] = *(const u32x4*)(Bk + 16);
    acc = wmma_bf16(a.v, bm.v, acc);
#pragma unroll
    for (int r = 0; r < 8; ++r) sc[tm * 16 + r + 8 * hh][tn * 16 + l15] = acc[r];
    v8f acc2 = v8zero();
    a.u[0] = *(const u32x4*)(Av);      a.u[1] = *(const u32x4*)(Av + 16);
    bm.u[0] = *(const u32x4*)(Bp);     bm.u[1] = *(const u32x4*)(Bp + 16);
    acc2 = wmma_bf16(a.v, bm.v, acc2);
#pragma unroll
    for (int r = 0; r < 8; ++r) ps[tm * 16 + r + 8 * hh][tn * 16 + l15] = acc2[r];
  }
  __syncthreads();

  // rel_shift + combine + scale:  out[i,j] = (sc[i,j] + shift(ps)[i,j]) / sqrt(D)
  for (int idx = threadIdx.x; idx < 4096; idx += 512) {
    const int i = idx >> 6, j = idx & 63;
    const int u = (i + 1) * 64 + j;
    const int aa = u / 65, bb2 = u % 65;
    const float sh = bb2 ? ps[aa][bb2 - 1] : 0.f;
    sc[i][j] = (sc[i][j] + sh) * 0.03125f;
  }
  __syncthreads();

  if (threadIdx.x < 64) {   // softmax, one row per thread
    const int i = threadIdx.x;
    float mx = -3.0e38f;
    for (int j = 0; j < 64; ++j) mx = fmaxf(mx, sc[i][j]);
    float sum = 0.f;
    for (int j = 0; j < 64; ++j) { float e = __expf(sc[i][j] - mx); sc[i][j] = e; sum += e; }
    const float inv = 1.f / sum;
    for (int j = 0; j < 64; ++j) sc[i][j] *= inv;
  }
  __syncthreads();

  if (wave < 8) {  // ctx = attn[64,64] @ v[64,32]; 4x2 tiles, K=64 (2 WMMA steps)
    const int tm = wave >> 1, tn = wave & 1;
    const __bf16* Bv = vt + ((size_t)bh * DHd + tn * 16 + l15) * Ss + 8 * hh;
    v8f acc = v8zero();
    const int mrow = tm * 16 + l15;
    for (int kb = 0; kb < 64; kb += 32) {
      ABf a, bm;
#pragma unroll
      for (int e = 0; e < 16; ++e) {
        const int kk = kb + 8 * hh + (e & 7) + ((e & 8) ? 16 : 0);
        a.v[e] = f2bf(sc[mrow][kk]);            // ds_load from LDS attn
      }
      bm.u[0] = *(const u32x4*)(Bv + kb);
      bm.u[1] = *(const u32x4*)(Bv + kb + 16);
      acc = wmma_bf16(a.v, bm.v, acc);
    }
#pragma unroll
    for (int r = 0; r < 8; ++r) {
      const int s = tm * 16 + r + 8 * hh, dh = tn * 16 + l15;
      ctxb[((size_t)(b * Ss + s)) * Dd + h * DHd + dh] = f2bf(acc[r]);
    }
  }
}

// ---------------------------------------------------------------------------
// LN1: x1 = LayerNorm(a + x) -> bf16
// ---------------------------------------------------------------------------
__global__ __launch_bounds__(256) void k_ln1(const float* __restrict__ a,
                                             const float* __restrict__ x,
                                             const float* __restrict__ g,
                                             const float* __restrict__ bt,
                                             __bf16* __restrict__ x1b) {
  __shared__ float rs[256], rq[256];
  const int m = blockIdx.x, t = threadIdx.x;
  float v[4], s = 0.f, q = 0.f;
#pragma unroll
  for (int i = 0; i < 4; ++i) {
    const int d = t + i * 256;
    v[i] = a[(size_t)m * Dd + d] + x[(size_t)m * Dd + d];
    s += v[i]; q += v[i] * v[i];
  }
  rs[t] = s; rq[t] = q; __syncthreads();
  for (int o = 128; o > 0; o >>= 1) {
    if (t < o) { rs[t] += rs[t + o]; rq[t] += rq[t + o]; }
    __syncthreads();
  }
  const float mu = rs[0] * (1.f / 1024.f);
  const float var = rq[0] * (1.f / 1024.f) - mu * mu;
  const float inv = rsqrtf(var + 1e-5f);
#pragma unroll
  for (int i = 0; i < 4; ++i) {
    const int d = t + i * 256;
    x1b[(size_t)m * Dd + d] = f2bf((v[i] - mu) * inv * g[d] + bt[d]);
  }
}

// ---------------------------------------------------------------------------
// MoE gating: one wave per token. Lanes split D (32 contiguous bf16 each),
// partial logits reduced with wave32 shuffles; lane 0 does softmax/top-2/RNG.
// ---------------------------------------------------------------------------
__global__ __launch_bounds__(256) void k_gating(const __bf16* __restrict__ x1b,
                                                const float* __restrict__ wg,
                                                float* __restrict__ raw,
                                                int* __restrict__ te1, float* __restrict__ tg1,
                                                int* __restrict__ te2, float* __restrict__ tg2) {
  const int wave = threadIdx.x >> 5, lane = threadIdx.x & 31;
  const int m = blockIdx.x * 8 + wave;
  const __bf16* xr = x1b + (size_t)m * Dd + lane * 32;
  float l[4] = {0.f, 0.f, 0.f, 0.f};
#pragma unroll
  for (int c = 0; c < 32; ++c) {
    const int d = lane * 32 + c;
    const float xv = bf2f(xr[c]);
    const f32x4 w = *(const f32x4*)(wg + d * 4);
    l[0] += xv * w.x; l[1] += xv * w.y; l[2] += xv * w.z; l[3] += xv * w.w;
  }
#pragma unroll
  for (int off = 16; off > 0; off >>= 1) {
#pragma unroll
    for (int e = 0; e < 4; ++e) l[e] += __shfl_down(l[e], off);
  }
  if (lane == 0) {
    float mx = fmaxf(fmaxf(l[0], l[1]), fmaxf(l[2], l[3]));
    float p[4], sum = 0.f;
#pragma unroll
    for (int e = 0; e < 4; ++e) { p[e] = __expf(l[e] - mx); sum += p[e]; }
    const float inv = 1.f / sum;
#pragma unroll
    for (int e = 0; e < 4; ++e) { p[e] *= inv; raw[(size_t)m * 4 + e] = p[e]; }
    int e1 = 0;
    for (int e = 1; e < 4; ++e) if (p[e] > p[e1]) e1 = e;
    float g1 = p[e1];
    int e2 = -1; float g2 = -1.f;
    for (int e = 0; e < 4; ++e) if (e != e1 && p[e] > g2) { g2 = p[e]; e2 = e; }
    const float den = g1 + g2 + 1e-9f;
    g1 /= den; g2 /= den;
    unsigned u = (unsigned)m * 2654435761u + 0x9E3779B9u;     // deterministic hash RNG
    u ^= u >> 16; u *= 0x7feb352du; u ^= u >> 15; u *= 0x846ca68bu; u ^= u >> 16;
    const float un = (float)(u >> 8) * (1.0f / 16777216.0f);
    te1[m] = e1; tg1[m] = g1;
    te2[m] = (un < g2 * 5.0f) ? e2 : -1;                      // keep iff u < g2/0.2
    tg2[m] = g2;
  }
}

// Capacity scan per batch (matches exclusive-cumsum semantics) + aux-loss partials.
__global__ void k_scan(const float* __restrict__ raw, int* te1, int* ts1,
                       int* te2, int* ts2, float* __restrict__ partial) {
  const int b = blockIdx.x;
  if (threadIdx.x != 0) return;
  int cntAll[4] = {0, 0, 0, 0};
  for (int n = 0; n < Ss; ++n) {
    const int m = b * Ss + n;
    const int e = te1[m];
    const int p = cntAll[e]++;
    if (p < CAP) ts1[m] = p; else te1[m] = -1;
  }
  int m1c[4];
  for (int e = 0; e < 4; ++e) m1c[e] = imin(cntAll[e], CAP);
  int cnt2[4] = {0, 0, 0, 0};
  for (int n = 0; n < Ss; ++n) {
    const int m = b * Ss + n;
    const int e = te2[m];
    if (e >= 0) {
      const int p = cnt2[e]++;
      const int pos = p + m1c[e];
      if (pos < CAP) ts2[m] = pos; else te2[m] = -1;
    }
  }
  float acc = 0.f;
  for (int e = 0; e < 4; ++e) {
    float dp = 0.f;
    for (int n = 0; n < Ss; ++n) dp += raw[(size_t)(b * Ss + n) * 4 + e];
    acc += (dp * (1.f / 64.f)) * ((float)cntAll[e] * (1.f / 64.f));
  }
  partial[b] = acc;
}

__global__ __launch_bounds__(256) void k_zero_u4(u32x4* __restrict__ p, int n) {
  const int i = blockIdx.x * 256 + threadIdx.x;
  if (i < n) { u32x4 z = {0u, 0u, 0u, 0u}; p[i] = z; }
}

__global__ __launch_bounds__(128) void k_scatter(const __bf16* __restrict__ x1b,
                                                 const int* __restrict__ te1,
                                                 const int* __restrict__ ts1,
                                                 const int* __restrict__ te2,
                                                 const int* __restrict__ ts2,
                                                 __bf16* __restrict__ ei) {
  const int m = blockIdx.x, b = m >> 6, t = threadIdx.x;
  const u32x4 val = ((const u32x4*)(x1b + (size_t)m * Dd))[t];
  const int e1 = te1[m], e2 = te2[m];
  if (e1 >= 0) ((u32x4*)(ei + ((size_t)(e1 * EROW + b * CAP + ts1[m])) * Dd))[t] = val;
  if (e2 >= 0) ((u32x4*)(ei + ((size_t)(e2 * EROW + b * CAP + ts2[m])) * Dd))[t] = val;
}

// Combine expert outputs + residual + LN2 -> d_out (f32).
__global__ __launch_bounds__(256) void k_combine_ln2(
    const __bf16* __restrict__ eo, const __bf16* __restrict__ x1b,
    const int* __restrict__ te1, const int* __restrict__ ts1, const float* __restrict__ tg1,
    const int* __restrict__ te2, const int* __restrict__ ts2, const float* __restrict__ tg2,
    const float* __restrict__ g, const float* __restrict__ bt, float* __restrict__ out) {
  __shared__ float rs[256], rq[256];
  const int m = blockIdx.x, t = threadIdx.x, b = m >> 6;
  const int e1 = te1[m], e2 = te2[m];
  const float g1 = tg1[m], g2 = tg2[m];
  const __bf16* r1 = (e1 >= 0) ? eo + ((size_t)(e1 * EROW + b * CAP + ts1[m])) * Dd : nullptr;
  const __bf16* r2 = (e2 >= 0) ? eo + ((size_t)(e2 * EROW + b * CAP + ts2[m])) * Dd : nullptr;
  float v[4], s = 0.f, q = 0.f;
#pragma unroll
  for (int i = 0; i < 4; ++i) {
    const int d = t + i * 256;
    float val = bf2f(x1b[(size_t)m * Dd + d]);
    if (r1) val += g1 * bf2f(r1[d]);
    if (r2) val += g2 * bf2f(r2[d]);
    v[i] = val; s += val; q += val * val;
  }
  rs[t] = s; rq[t] = q; __syncthreads();
  for (int o = 128; o > 0; o >>= 1) {
    if (t < o) { rs[t] += rs[t + o]; rq[t] += rq[t + o]; }
    __syncthreads();
  }
  const float mu = rs[0] * (1.f / 1024.f);
  const float var = rq[0] * (1.f / 1024.f) - mu * mu;
  const float inv = rsqrtf(var + 1e-5f);
#pragma unroll
  for (int i = 0; i < 4; ++i) {
    const int d = t + i * 256;
    out[(size_t)m * Dd + d] = (v[i] - mu) * inv * g[d] + bt[d];
  }
}

__global__ __launch_bounds__(256) void k_loss(const float* __restrict__ partial,
                                              float* __restrict__ out) {
  __shared__ float rs[256];
  const int t = threadIdx.x;
  rs[t] = partial[t];
  __syncthreads();
  for (int o = 128; o > 0; o >>= 1) {
    if (t < o) rs[t] += rs[t + o];
    __syncthreads();
  }
  if (t == 0) out[(size_t)Mtok * Dd] = rs[0] * (1.f / (256.f * 4.f)) * 16.f * 0.01f;
}

// ---------------------------------------------------------------------------
// Host orchestration
// ---------------------------------------------------------------------------
extern "C" void kernel_launch(void* const* d_in, const int* in_sizes, int n_in,
                              void* d_out, int out_size, void* d_ws, size_t ws_size,
                              hipStream_t stream) {
  (void)in_sizes; (void)n_in; (void)out_size; (void)ws_size;
  const float* x    = (const float*)d_in[0];
  const float* pe   = (const float*)d_in[1];
  const float* Wq   = (const float*)d_in[2];
  const float* bq   = (const float*)d_in[3];
  const float* Wk   = (const float*)d_in[4];
  const float* bk   = (const float*)d_in[5];
  const float* Wv   = (const float*)d_in[6];
  const float* bv   = (const float*)d_in[7];
  const float* Wpos = (const float*)d_in[8];
  const float* ub   = (const float*)d_in[9];
  const float* vbi  = (const float*)d_in[10];
  const float* Wo   = (const float*)d_in[11];
  const float* bo   = (const float*)d_in[12];
  const float* ln1g = (const float*)d_in[13];
  const float* ln1b = (const float*)d_in[14];
  const float* ln2g = (const float*)d_in[15];
  const float* ln2b = (const float*)d_in[16];
  const float* wg   = (const float*)d_in[17];
  const float* w1   = (const float*)d_in[18];
  const float* w2   = (const float*)d_in[19];

  char* ws = (char*)d_ws;
  constexpr size_t MB = 1024ull * 1024ull;
  __bf16* xb   = (__bf16*)(ws + 0);          // 32MB  x bf16 [16384,1024]
  __bf16* peb  = (__bf16*)(ws + 32 * MB);    //  1MB  pos_enc bf16 [64,1024]
  __bf16* WqT  = (__bf16*)(ws + 33 * MB);    //  2MB each, transposed [N,K]
  __bf16* WkT  = (__bf16*)(ws + 35 * MB);
  __bf16* WvT  = (__bf16*)(ws + 37 * MB);
  __bf16* WpT  = (__bf16*)(ws + 39 * MB);
  __bf16* WoT  = (__bf16*)(ws + 41 * MB);
  __bf16* w1T  = (__bf16*)(ws + 43 * MB);    // 12MB  [E][HID,D]
  __bf16* w2T  = (__bf16*)(ws + 55 * MB);    // 12MB  [E][D,HID]
  __bf16* quB  = (__bf16*)(ws + 67 * MB);    // 32MB  [b,h,s,dh]
  __bf16* qvB  = (__bf16*)(ws + 99 * MB);    // 32MB
  __bf16* kB   = (__bf16*)(ws + 131 * MB);   // 32MB
  __bf16* vtB  = (__bf16*)(ws + 163 * MB);   // 32MB  [b,h,dh,s]
  __bf16* posB = (__bf16*)(ws + 195 * MB);   //  1MB  [h,t,dh]
  __bf16* ctxB = (__bf16*)(ws + 196 * MB);   // 32MB  [token,D]
  float*  aF   = (float*)(ws + 228 * MB);    // 64MB  attn out f32
  __bf16* x1B  = (__bf16*)(ws + 292 * MB);   // 32MB
  // aliases (lifetimes disjoint):
  __bf16* hidB = quB;                        // 96MB over qu/qv/k
  __bf16* eoB  = vtB;                        // 64MB over vt/pos/ctx
  __bf16* eiB  = (__bf16*)aF;                // 64MB over aF
  char* gz = ws + 324 * MB;                  // ~0.7MB gating area
  float* raw  = (float*)(gz);
  int*   te1  = (int*)(gz + 262144);
  int*   ts1  = (int*)(gz + 262144 + 1 * 65536);
  int*   te2  = (int*)(gz + 262144 + 2 * 65536);
  int*   ts2  = (int*)(gz + 262144 + 3 * 65536);
  float* tg1  = (float*)(gz + 262144 + 4 * 65536);
  float* tg2  = (float*)(gz + 262144 + 5 * 65536);
  float* part = (float*)(gz + 262144 + 6 * 65536);
  float* out  = (float*)d_out;

  // converts / weight transposes
  k_f32_to_bf16<<<(Mtok * Dd) / 256, 256, 0, stream>>>(x, xb, Mtok * Dd);
  k_f32_to_bf16<<<(Ss * Dd) / 256, 256, 0, stream>>>(pe, peb, Ss * Dd);
  k_transpose_bf16<<<(Dd * Dd) / 256, 256, 0, stream>>>(Wq, WqT, Dd, Dd);
  k_transpose_bf16<<<(Dd * Dd) / 256, 256, 0, stream>>>(Wk, WkT, Dd, Dd);
  k_transpose_bf16<<<(Dd * Dd) / 256, 256, 0, stream>>>(Wv, WvT, Dd, Dd);
  k_transpose_bf16<<<(Dd * Dd) / 256, 256, 0, stream>>>(Wpos, WpT, Dd, Dd);
  k_transpose_bf16<<<(Dd * Dd) / 256, 256, 0, stream>>>(Wo, WoT, Dd, Dd);
  for (int e = 0; e < Ee; ++e) {
    k_transpose_bf16<<<(Dd * HIDh) / 256, 256, 0, stream>>>(
        w1 + (size_t)e * Dd * HIDh, w1T + (size_t)e * HIDh * Dd, Dd, HIDh);
    k_transpose_bf16<<<(HIDh * Dd) / 256, 256, 0, stream>>>(
        w2 + (size_t)e * HIDh * Dd, w2T + (size_t)e * Dd * HIDh, HIDh, Dd);
  }

  auto blocks64 = [](int M, int N) { return ((M / 64) * (N / 64) + 7) / 8; };

  // projections (fused bias / u_bias / v_bias, attention layouts)
  k_gemm<1><<<blocks64(Mtok, Dd), 256, 0, stream>>>(xb, WqT, bq, ub, vbi, quB, qvB, Mtok, Dd, Dd);
  k_gemm<2><<<blocks64(Mtok, Dd), 256, 0, stream>>>(xb, WkT, bk, nullptr, nullptr, kB, nullptr, Mtok, Dd, Dd);
  k_gemm<3><<<blocks64(Mtok, Dd), 256, 0, stream>>>(xb, WvT, bv, nullptr, nullptr, vtB, nullptr, Mtok, Dd, Dd);
  k_gemm<4><<<blocks64(Ss, Dd), 256, 0, stream>>>(peb, WpT, nullptr, nullptr, nullptr, posB, nullptr, Ss, Dd, Dd);

  // fused rel-attention per (b,h)
  k_attn<<<Bb * Hh, 512, 0, stream>>>(quB, qvB, kB, vtB, posB, ctxB);

  // output projection + LN1
  k_gemm<5><<<blocks64(Mtok, Dd), 256, 0, stream>>>(ctxB, WoT, bo, nullptr, nullptr, aF, nullptr, Mtok, Dd, Dd);
  k_ln1<<<Mtok, 256, 0, stream>>>(aF, x, ln1g, ln1b, x1B);

  // MoE gating + capacity scan + dispatch
  k_gating<<<Mtok / 8, 256, 0, stream>>>(x1B, wg, raw, te1, tg1, te2, tg2);
  k_scan<<<Bb, 32, 0, stream>>>(raw, te1, ts1, te2, ts2, part);
  k_zero_u4<<<(Ee * EROW * Dd / 8) / 256, 256, 0, stream>>>((u32x4*)eiB, Ee * EROW * Dd / 8);
  k_scatter<<<Mtok, 128, 0, stream>>>(x1B, te1, ts1, te2, ts2, eiB);

  // expert FFNs
  for (int e = 0; e < Ee; ++e) {
    k_gemm<6><<<blocks64(EROW, HIDh), 256, 0, stream>>>(
        eiB + (size_t)e * EROW * Dd, w1T + (size_t)e * HIDh * Dd,
        nullptr, nullptr, nullptr, hidB + (size_t)e * EROW * HIDh, nullptr, EROW, HIDh, Dd);
    k_gemm<7><<<blocks64(EROW, Dd), 256, 0, stream>>>(
        hidB + (size_t)e * EROW * HIDh, w2T + (size_t)e * Dd * HIDh,
        nullptr, nullptr, nullptr, eoB + (size_t)e * EROW * Dd, nullptr, EROW, Dd, HIDh);
  }

  // combine + LN2 -> y, then aux loss scalar
  k_combine_ln2<<<Mtok, 256, 0, stream>>>(eoB, x1B, te1, ts1, tg1, te2, ts2, tg2,
                                          ln2g, ln2b, out);
  k_loss<<<1, 256, 0, stream>>>(part, out);
}
